// PolicyNet_59459527246286
// MI455X (gfx1250) — matrix operand order repaired and checked by reading, not compiled, not run on hardware
//
#include <hip/hip_runtime.h>

typedef __attribute__((ext_vector_type(16))) _Float16 v16h;
typedef __attribute__((ext_vector_type(8)))  _Float16 v8h;
typedef __attribute__((ext_vector_type(8)))  float    v8f;
typedef __attribute__((ext_vector_type(4)))  float    v4f;

#define THREADS 256
#define ROWS_PER_BLOCK 128

// LDS strides (in halves), padded so row/col stride mod 64 banks steps by 4
// banks per row -> conflict-free b128 loads across 16 lanes.
#define S_IN 104   // input rows, K padded 65->96 (read up to 95)
#define S_W1 104   // W1 [128][65] padded
#define S_A1 136   // act1 [128][128]
#define S_W2 136   // W2 [64][128]
#define S_A2 72    // act2 [128][64]
#define S_W3 72    // W3 [256][64]

// byte offsets into dynamic LDS (total 163072 -> 163840 after 2KB rounding,
// exactly half of the 320KB WGP LDS => 2 blocks resident per WGP)
#define OFF_W1 0
#define OFF_W2 26624    // 128*104*2
#define OFF_W3 44032    // +64*136*2
#define OFF_IN 80896    // +256*72*2
#define OFF_A1 107520   // +128*104*2
#define OFF_A2 142336   // +128*136*2
#define OFF_B1 160768   // +128*72*2
#define OFF_B2 161280
#define OFF_B3 161536
#define OFF_SC 162560
#define LDS_BYTES 163072

// A operand (16x32 f16, MxK): lane L holds row M=L&15.
// v0..3 = K[kbase + (L>=16?8:0) .. +7], v4..7 = same +16.  Two b128 LDS loads.
__device__ __forceinline__ v16h load_A(const _Float16* base, int stride, int kbase, int lane) {
    int row = lane & 15;
    int ko  = kbase + ((lane >> 4) << 3);
    const _Float16* p = base + row * stride + ko;
    v16h r;
    *(v8h*)&r       = *(const v8h*)p;
    *((v8h*)&r + 1) = *(const v8h*)(p + 16);
    return r;
}

// B operand (32x16 f16, KxN): lane L holds col N=n0+(L&15),
// v0..3 = K[kbase + (L>=16?8:0) .. +7], v4..7 = same +16.  Two b128 LDS loads.
// Weights stored [N][K] row-major (== W as given, since W is (out,in)).
__device__ __forceinline__ v16h load_B(const _Float16* base, int stride, int n0, int kbase, int lane) {
    int col = n0 + (lane & 15);
    int ko  = kbase + ((lane >> 4) << 3);
    const _Float16* p = base + col * stride + ko;
    v16h r;
    *(v8h*)&r       = *(const v8h*)p;
    *((v8h*)&r + 1) = *(const v8h*)(p + 16);
    return r;
}

__global__ __launch_bounds__(THREADS)
void policy_mlp_kernel(const float* __restrict__ inp,
                       const int*   __restrict__ scale_table,
                       const float* __restrict__ W1, const float* __restrict__ b1,
                       const float* __restrict__ W2, const float* __restrict__ b2,
                       const float* __restrict__ W3, const float* __restrict__ b3,
                       float* __restrict__ outP)
{
    extern __shared__ char smem[];
    _Float16* w1h   = (_Float16*)(smem + OFF_W1);
    _Float16* w2h   = (_Float16*)(smem + OFF_W2);
    _Float16* w3h   = (_Float16*)(smem + OFF_W3);
    _Float16* actin = (_Float16*)(smem + OFF_IN);
    _Float16* act1  = (_Float16*)(smem + OFF_A1);
    _Float16* act2  = (_Float16*)(smem + OFF_A2);
    float*    b1s   = (float*)(smem + OFF_B1);
    float*    b2s   = (float*)(smem + OFF_B2);
    float*    b3s   = (float*)(smem + OFF_B3);
    int*      sScl  = (int*)  (smem + OFF_SC);

    const int tid   = threadIdx.x;
    const int lane  = tid & 31;
    const int wave  = tid >> 5;
    const int rbase = blockIdx.x * ROWS_PER_BLOCK;
    const int lcol  = lane & 15;
    const int lrow  = (lane >> 4) << 3;   // 0 or 8 (C-layout M split)

    // ---- phase 0: stage weights/biases/inputs into LDS as f16 ----
    for (int i = tid; i < 128 * S_W1; i += THREADS) {
        int r = i / S_W1, c = i - r * S_W1;
        w1h[i] = (_Float16)((c < 65) ? W1[r * 65 + c] : 0.f);
    }
    for (int i = tid; i < 64 * S_W2; i += THREADS) {
        int r = i / S_W2, c = i - r * S_W2;
        w2h[i] = (_Float16)((c < 128) ? W2[r * 128 + c] : 0.f);
    }
    for (int i = tid; i < 256 * S_W3; i += THREADS) {
        int r = i / S_W3, c = i - r * S_W3;
        w3h[i] = (_Float16)((c < 64) ? W3[r * 64 + c] : 0.f);
    }
    for (int i = tid; i < ROWS_PER_BLOCK * S_IN; i += THREADS) {
        int r = i / S_IN, c = i - r * S_IN;
        actin[i] = (_Float16)((c < 65) ? inp[(rbase + r) * 65 + c] : 0.f);
    }
    if (tid < 128) b1s[tid] = b1[tid];
    if (tid < 64)  b2s[tid] = b2[tid];
    b3s[tid] = b3[tid];
    if (tid < ROWS_PER_BLOCK) {
        int di = (int)inp[(rbase + tid) * 65 + 64];
        sScl[tid] = scale_table[di & 15];
    }
    __syncthreads();

    const int m0 = wave * 16;   // this wave's 16-row stripe

    // ---- layer 1: (16x65) x (65x128), K padded to 96 ----
    #pragma unroll
    for (int n = 0; n < 8; ++n) {
        v8f acc = {};
        #pragma unroll
        for (int k = 0; k < 3; ++k) {
            v16h a  = load_A(actin + m0 * S_IN, S_IN, k * 32, lane);
            v16h bm = load_B(w1h, S_W1, n * 16, k * 32, lane);
            acc = __builtin_amdgcn_wmma_f32_16x16x32_f16(false, a, false, bm,
                                                         (short)0, acc, false, false);
        }
        float bv = b1s[n * 16 + lcol];
        #pragma unroll
        for (int r = 0; r < 8; ++r) {
            int row = m0 + r + lrow;
            act1[row * S_A1 + n * 16 + lcol] = (_Float16)fmaxf(acc[r] + bv, 0.f);
        }
    }
    // intra-wave LDS RAW is in-order (DScnt); barrier kept for safety, cheap
    __syncthreads();

    // ---- layer 2: (16x128) x (128x64) ----
    #pragma unroll
    for (int n = 0; n < 4; ++n) {
        v8f acc = {};
        #pragma unroll
        for (int k = 0; k < 4; ++k) {
            v16h a  = load_A(act1 + m0 * S_A1, S_A1, k * 32, lane);
            v16h bm = load_B(w2h, S_W2, n * 16, k * 32, lane);
            acc = __builtin_amdgcn_wmma_f32_16x16x32_f16(false, a, false, bm,
                                                         (short)0, acc, false, false);
        }
        float bv = b2s[n * 16 + lcol];
        #pragma unroll
        for (int r = 0; r < 8; ++r) {
            int row = m0 + r + lrow;
            act2[row * S_A2 + n * 16 + lcol] = (_Float16)fmaxf(acc[r] + bv, 0.f);
        }
    }
    __syncthreads();

    // ---- layer 3: (16x64) x (64x256), logits stay in registers ----
    v8f acc3[16];
    #pragma unroll
    for (int n = 0; n < 16; ++n) {
        v8f acc = {};
        #pragma unroll
        for (int k = 0; k < 2; ++k) {
            v16h a  = load_A(act2 + m0 * S_A2, S_A2, k * 32, lane);
            v16h bm = load_B(w3h, S_W3, n * 16, k * 32, lane);
            acc = __builtin_amdgcn_wmma_f32_16x16x32_f16(false, a, false, bm,
                                                         (short)0, acc, false, false);
        }
        acc3[n] = acc;
    }

    // ---- masked softmax over 256 logits per row, in registers ----
    // VGPR r of tile t holds logits[row = m0 + r + lrow][col = t*16 + lcol]
    int   scl[8];
    float rowmax[8], rsum[8], inv[8];
    #pragma unroll
    for (int r = 0; r < 8; ++r) {
        scl[r]    = sScl[m0 + r + lrow];
        rowmax[r] = -1e30f;
    }
    #pragma unroll
    for (int t = 0; t < 16; ++t) {
        int   col = t * 16 + lcol;
        float bv  = b3s[col];
        #pragma unroll
        for (int r = 0; r < 8; ++r) {
            float v = acc3[t][r] + bv;
            if (col >= scl[r]) v = -1e30f;      // mask (matches -1e9 => prob 0)
            acc3[t][r] = v;
            rowmax[r]  = fmaxf(rowmax[r], v);
        }
    }
    #pragma unroll
    for (int r = 0; r < 8; ++r) {               // reduce across the 16-lane half
        float m = rowmax[r];
        m = fmaxf(m, __shfl_xor(m, 1, 16));
        m = fmaxf(m, __shfl_xor(m, 2, 16));
        m = fmaxf(m, __shfl_xor(m, 4, 16));
        m = fmaxf(m, __shfl_xor(m, 8, 16));
        rowmax[r] = m;
        rsum[r] = 0.f;
    }
    #pragma unroll
    for (int t = 0; t < 16; ++t)
        #pragma unroll
        for (int r = 0; r < 8; ++r) {
            float e = __expf(acc3[t][r] - rowmax[r]);
            acc3[t][r] = e;
            rsum[r] += e;
        }
    #pragma unroll
    for (int r = 0; r < 8; ++r) {
        float s = rsum[r];
        s += __shfl_xor(s, 1, 16);
        s += __shfl_xor(s, 2, 16);
        s += __shfl_xor(s, 4, 16);
        s += __shfl_xor(s, 8, 16);
        inv[r] = 1.0f / s;
    }
    // streaming (non-temporal) stores: 256MB write-once, keep out of L2
    #pragma unroll
    for (int t = 0; t < 16; ++t) {
        int col = t * 16 + lcol;
        #pragma unroll
        for (int r = 0; r < 8; ++r) {
            int grow = rbase + m0 + r + lrow;
            __builtin_nontemporal_store(acc3[t][r] * inv[r],
                                        outP + (size_t)grow * 256 + col);
        }
    }
}

// mask output: pure bandwidth, independent of the MLP -> own kernel, b128 NT stores
__global__ __launch_bounds__(THREADS)
void mask_kernel(const float* __restrict__ inp, const int* __restrict__ scale_table,
                 float* __restrict__ outM)
{
    int tid = blockIdx.x * blockDim.x + threadIdx.x;   // 64 threads per row
    int row = tid >> 6;
    int c0  = (tid & 63) << 2;
    int di  = (int)inp[row * 65 + 64];
    int sc  = scale_table[di & 15];
    v4f v;
    v.x = (c0 + 0 < sc) ? 1.f : 0.f;
    v.y = (c0 + 1 < sc) ? 1.f : 0.f;
    v.z = (c0 + 2 < sc) ? 1.f : 0.f;
    v.w = (c0 + 3 < sc) ? 1.f : 0.f;
    __builtin_nontemporal_store(v, (v4f*)(outM + (size_t)row * 256 + c0));
}

extern "C" void kernel_launch(void* const* d_in, const int* in_sizes, int n_in,
                              void* d_out, int out_size, void* d_ws, size_t ws_size,
                              hipStream_t stream) {
    const float* inp = (const float*)d_in[0];
    const int*   st  = (const int*)d_in[1];
    const float* W1  = (const float*)d_in[2];
    const float* b1  = (const float*)d_in[3];
    const float* W2  = (const float*)d_in[4];
    const float* b2  = (const float*)d_in[5];
    const float* W3  = (const float*)d_in[6];
    const float* b3  = (const float*)d_in[7];

    int Bn = in_sizes[0] / 65;                 // 262144
    float* outP = (float*)d_out;
    float* outM = outP + (size_t)Bn * 256;

    // idempotent host-side attribute set; not a stream op, capture-safe
    (void)hipFuncSetAttribute((const void*)policy_mlp_kernel,
                              hipFuncAttributeMaxDynamicSharedMemorySize, LDS_BYTES);

    policy_mlp_kernel<<<Bn / ROWS_PER_BLOCK, THREADS, LDS_BYTES, stream>>>(
        inp, st, W1, b1, W2, b2, W3, b3, outP);
    mask_kernel<<<(Bn * 64) / THREADS, THREADS, 0, stream>>>(inp, st, outM);
}
